// TemporalGATAttention_8821862826027
// MI455X (gfx1250) — matrix operand order repaired and checked by reading, not compiled, not run on hardware
//
#include <hip/hip_runtime.h>
#include <hip/hip_bf16.h>

// Problem constants (match reference)
#define Bb 8
#define Cc 128
#define Tt 256
#define Nn 100
#define Hh 4
#define HDd 32
#define BN (Bb * Nn)   // 800
#define SLOPE 0.2f
#define EPSV 1e-5f

typedef __attribute__((ext_vector_type(16))) __bf16 v16bf;
typedef __attribute__((ext_vector_type(8)))  float  v8f;
typedef int v4i __attribute__((vector_size(16)));   // <4 x i32> for async-LDS builtin

#if __has_builtin(__builtin_amdgcn_global_load_async_to_lds_b128)
#define HAVE_ASYNC_LDS 1
#else
#define HAVE_ASYNC_LDS 0
#endif

// Native bf16 converts: gfx1250 has HW bf16 cvt (packed) — far cheaper than
// manual RNE bit twiddling.
__device__ __forceinline__ __bf16 f2bf(float f) { return (__bf16)f; }
__device__ __forceinline__ float  bf2f(__bf16 b) { return (float)b; }
__device__ __forceinline__ float lrelu(float v) { return v >= 0.f ? v : SLOPE * v; }

#if HAVE_ASYNC_LDS
__device__ __forceinline__ void async_b128(const void* g, void* l) {
    __builtin_amdgcn_global_load_async_to_lds_b128(
        (__attribute__((address_space(1))) v4i*)g,
        (__attribute__((address_space(3))) v4i*)l, 0, 0);
}
__device__ __forceinline__ void wait_async() {
    asm volatile("s_wait_asynccnt 0x0" ::: "memory");
}
#endif

// ---------------------------------------------------------------------------
// Kernel 0: fold attention vectors into projection weights (Q/K never
// materialized): u_src[h][c] = sum_d Wq[h*HD+d][c]*a_src[h][d]; same for dst.
// Also pre-convert Wv to bf16 (row-major) so kProj can async-copy raw bytes.
// ---------------------------------------------------------------------------
__global__ __launch_bounds__(128) void kPrep(const float* __restrict__ Wq,
                                             const float* __restrict__ Wk,
                                             const float* __restrict__ Wv,
                                             const float* __restrict__ as,
                                             const float* __restrict__ ad,
                                             float* __restrict__ u,
                                             __bf16* __restrict__ wvbf) {
    int c = threadIdx.x;  // 0..127
    for (int h = 0; h < Hh; ++h) {
        float s1 = 0.f, s2 = 0.f;
        for (int d = 0; d < HDd; ++d) {
            s1 += Wq[(h * HDd + d) * Cc + c] * as[h * HDd + d];
            s2 += Wk[(h * HDd + d) * Cc + c] * ad[h * HDd + d];
        }
        u[h * Cc + c]           = s1;  // u_src
        u[Hh * Cc + h * Cc + c] = s2;  // u_dst
    }
    for (int idx = threadIdx.x; idx < Cc * Cc; idx += 128)
        wvbf[idx] = f2bf(Wv[idx]);
}

// ---------------------------------------------------------------------------
// Kernel T: (B,C,T,N) fp32 -> (BN,T,C) bf16, LDS transpose so both global
// sides are coalesced. One block per (b,t).
// ---------------------------------------------------------------------------
__global__ __launch_bounds__(256) void kT(const float* __restrict__ x,
                                          __bf16* __restrict__ xr) {
    int b = blockIdx.x / Tt, t = blockIdx.x % Tt;
    __shared__ __bf16 tile[Cc][Nn];  // 25.6 KB
    for (int idx = threadIdx.x; idx < Cc * Nn; idx += 256) {
        int c = idx / Nn, n = idx % Nn;  // n fastest -> coalesced read
        tile[c][n] = f2bf(x[((size_t)(b * Cc + c) * Tt + t) * Nn + n]);
    }
    __syncthreads();
    for (int idx = threadIdx.x; idx < Cc * Nn / 2; idx += 256) {
        int c2 = (idx % (Cc / 2)) * 2, n = idx / (Cc / 2);  // c fastest -> coalesced write
        union { unsigned u; __bf16 h[2]; } p;
        p.h[0] = tile[c2][n]; p.h[1] = tile[c2 + 1][n];
        *(unsigned*)&xr[((size_t)(b * Nn + n) * Tt + t) * Cc + c2] = p.u;
    }
}

// ---------------------------------------------------------------------------
// Kernel 1: V projection (WMMA bf16) + s_src/s_dst row dots.
// One block = one bn x 64-row tile. 8 waves: wave w -> row-tile w%4,
// col-tiles (w/4)*4 .. +3. K-loop over C in 32-chunks.
// LDS fills go through GLOBAL_LOAD_ASYNC_TO_LDS_B128 when available.
// ---------------------------------------------------------------------------
__global__ __launch_bounds__(256) void kProj(const __bf16* __restrict__ xr,
                                             const __bf16* __restrict__ wvbf,
                                             const float* __restrict__ u,
                                             __bf16* __restrict__ Vw,
                                             float* __restrict__ ssrc,
                                             float* __restrict__ sdst) {
    int bn = blockIdx.x;
    int t0 = blockIdx.y * 64;
    // Row strides 136*2 = 272B (16B-aligned, bank-conflict free for WMMA reads)
    __shared__ __bf16 xs[64][Cc + 8];     // 17.4 KB
    __shared__ __bf16 wv[Cc][Cc + 8];     // wv[co][k] = Wv[co][k], 34.8 KB
    __shared__ float  us[Hh][Cc], ud[Hh][Cc];
    int tid = threadIdx.x;

    const __bf16* xsrc = xr + (size_t)(bn * Tt + t0) * Cc;
#if HAVE_ASYNC_LDS
    // 64 rows x 256B = 1024 x 16B chunks
    for (int idx = tid; idx < 64 * 16; idx += 256)
        async_b128(xsrc + idx * 8, &xs[idx >> 4][(idx & 15) * 8]);
    // 128 rows x 256B = 2048 x 16B chunks
    for (int idx = tid; idx < 128 * 16; idx += 256)
        async_b128(wvbf + idx * 8, &wv[idx >> 4][(idx & 15) * 8]);
#else
    for (int idx = tid; idx < 64 * Cc / 2; idx += 256) {
        int c2 = (idx * 2) & 127, r = (idx * 2) >> 7;
        *(unsigned*)&xs[r][c2] = ((const unsigned*)xsrc)[idx];
    }
    for (int idx = tid; idx < Cc * Cc / 2; idx += 256) {
        int c2 = (idx * 2) & 127, co = (idx * 2) >> 7;
        *(unsigned*)&wv[co][c2] = ((const unsigned*)wvbf)[idx];
    }
#endif
    for (int idx = tid; idx < Hh * Cc; idx += 256) {
        us[idx >> 7][idx & 127] = u[idx];
        ud[idx >> 7][idx & 127] = u[Hh * Cc + idx];
    }
#if HAVE_ASYNC_LDS
    wait_async();
#endif
    __syncthreads();

    // s_src / s_dst: thread -> (row r, head h)
    {
        int r = tid >> 2, h = tid & 3;
        float s1 = 0.f, s2 = 0.f;
        for (int c = 0; c < Cc; ++c) {
            float xv = bf2f(xs[r][c]);
            s1 += xv * us[h][c];
            s2 += xv * ud[h][c];
        }
        int t = t0 + r;
        ssrc[(size_t)(bn * Hh + h) * Tt + t] = s1;
        sdst[(size_t)(bn * Hh + h) * Tt + t] = s2;
    }

    // V = x @ Wv^T with v_wmma_f32_16x16x32_bf16
    int wave = tid >> 5, lane = tid & 31, g = lane >> 4, l16 = lane & 15;
    int rt = wave & 3;
    int ctBase = (wave >> 2) * 4;
    for (int j = 0; j < 4; ++j) {
        int ct = ctBase + j;
        v8f acc = {};
        for (int k0 = 0; k0 < Cc; k0 += 32) {
            v16bf a, bfr;
            int row = rt * 16 + l16;
            #pragma unroll
            for (int e = 0; e < 16; ++e) {   // contiguous k = 16*g + e (matches B)
                int K = g * 16 + e;
                a[e] = xs[row][k0 + K];
            }
            int co = ct * 16 + l16;
            #pragma unroll
            for (int e = 0; e < 16; ++e) {   // B[k][n=co] = Wv[co][k]
                bfr[e] = wv[co][k0 + g * 16 + e];
            }
            acc = __builtin_amdgcn_wmma_f32_16x16x32_bf16(
                false, a, false, bfr, (short)0, acc, false, false);
        }
        #pragma unroll
        for (int r = 0; r < 8; ++r) {        // C/D layout: M = r + 8*g
            int t = t0 + rt * 16 + r + 8 * g;
            Vw[(size_t)(bn * Tt + t) * Cc + ct * 16 + l16] = f2bf(acc[r]);
        }
    }
}

// ---------------------------------------------------------------------------
// Kernel 2: causal additive-attention softmax + P@V_h via WMMA.
// One block per (bn, h). leaky_relu is monotone increasing, so
// rowmax[t] = lrelu(s_src[t] + cummax(s_dst)[0..t]).
// Probabilities are synthesized per-lane directly in the WMMA A layout.
// V tile stored transposed (VhT[co][k]) so each B-fragment is 16 contiguous
// bf16 -> ds_load_b128.
// ---------------------------------------------------------------------------
__global__ __launch_bounds__(256) void kAttn(const __bf16* __restrict__ Vw,
                                             const float* __restrict__ ssrcG,
                                             const float* __restrict__ sdstG,
                                             float* __restrict__ outw) {
    int bn = blockIdx.x, h = blockIdx.y;
    __shared__ __bf16 VhT[HDd][Tt + 8];   // 16.9 KB, row stride 528B
    __shared__ float ss[Tt], sd[Tt], cm[Tt], mrow[Tt], invd[Tt];
    int tid = threadIdx.x;

    for (int idx = tid; idx < Tt * HDd / 2; idx += 256) {
        int t = idx >> 4, c2 = (idx & 15) * 2;   // coalesced global read
        union { unsigned u; __bf16 h2[2]; } p;
        p.u = *(const unsigned*)&Vw[(size_t)(bn * Tt + t) * Cc + h * HDd + c2];
        VhT[c2][t]     = p.h2[0];
        VhT[c2 + 1][t] = p.h2[1];
    }
    ss[tid] = ssrcG[(size_t)(bn * Hh + h) * Tt + tid];
    sd[tid] = sdstG[(size_t)(bn * Hh + h) * Tt + tid];
    __syncthreads();

    // inclusive max-scan of sd -> cm (Hillis-Steele)
    cm[tid] = sd[tid];
    __syncthreads();
    for (int off = 1; off < Tt; off <<= 1) {
        float v = cm[tid];
        float o = (tid >= off) ? cm[tid - off] : v;
        __syncthreads();
        cm[tid] = fmaxf(v, o);
        __syncthreads();
    }

    // per-row max + denom (thread t owns row t)
    {
        int t = tid;
        float m = lrelu(ss[t] + cm[t]);
        float d = 0.f;
        for (int s = 0; s <= t; ++s) d += __expf(lrelu(ss[t] + sd[s]) - m);
        mrow[t] = m;
        invd[t] = 1.f / d;
    }
    __syncthreads();

    // P @ V_h: wave handles row-tiles {wave, wave+8}; causal K-chunk loop
    int wave = tid >> 5, lane = tid & 31, g = lane >> 4, l16 = lane & 15;
    for (int rr = 0; rr < 2; ++rr) {
        int rt = wave + rr * 8;
        int t0 = rt * 16;
        v8f acc0 = {}, acc1 = {};
        int rowA = t0 + l16;
        float mA = mrow[rowA], sA = ss[rowA];
        int nk = (rt >> 1) + 1;               // chunks covering s <= t0+15
        for (int kc = 0; kc < nk; ++kc) {
            int k0 = kc * 32;
            v16bf a, b0, b1;
            #pragma unroll
            for (int e = 0; e < 16; ++e) {    // probs into A, contiguous k = 16*g + e
                int K = g * 16 + e;
                int s = k0 + K;
                float p = (s <= rowA) ? __expf(lrelu(sA + sd[s]) - mA) : 0.f;
                a[e] = f2bf(p);
            }
            #pragma unroll
            for (int e = 0; e < 16; ++e) {    // contiguous -> ds_load_b128
                int k = k0 + g * 16 + e;
                b0[e] = VhT[l16][k];
                b1[e] = VhT[16 + l16][k];
            }
            acc0 = __builtin_amdgcn_wmma_f32_16x16x32_bf16(
                false, a, false, b0, (short)0, acc0, false, false);
            acc1 = __builtin_amdgcn_wmma_f32_16x16x32_bf16(
                false, a, false, b1, (short)0, acc1, false, false);
        }
        #pragma unroll
        for (int r = 0; r < 8; ++r) {
            int t = t0 + r + 8 * g;
            float sc = invd[t];
            size_t base = (size_t)(bn * Tt + t) * Cc + h * HDd;
            outw[base + l16]      = acc0[r] * sc;
            outw[base + 16 + l16] = acc1[r] * sc;
        }
    }
}

// ---------------------------------------------------------------------------
// Kernel 3: LayerNorm over C + transpose back to (B,C,T,N).
// One block per (b,t); wave-per-row mean/var via wave32 shuffles.
// ---------------------------------------------------------------------------
__global__ __launch_bounds__(256) void kLN(const float* __restrict__ outw,
                                           const float* __restrict__ gamma,
                                           const float* __restrict__ beta,
                                           float* __restrict__ y) {
    int b = blockIdx.x / Tt, t = blockIdx.x % Tt;
    __shared__ float tile[Cc][Nn];       // 51.2 KB
    int wave = threadIdx.x >> 5, lane = threadIdx.x & 31;
    for (int n = wave; n < Nn; n += 8) {
        const float* row = outw + ((size_t)(b * Nn + n) * Tt + t) * Cc;
        float v[4], s = 0.f, s2 = 0.f;
        #pragma unroll
        for (int i = 0; i < 4; ++i) {
            v[i] = row[lane + 32 * i];
            s += v[i]; s2 += v[i] * v[i];
        }
        #pragma unroll
        for (int o = 16; o >= 1; o >>= 1) {
            s  += __shfl_xor(s,  o, 32);
            s2 += __shfl_xor(s2, o, 32);
        }
        float mu  = s / Cc;
        float var = s2 / Cc - mu * mu;
        float rstd = rsqrtf(var + EPSV);
        #pragma unroll
        for (int i = 0; i < 4; ++i) {
            int c = lane + 32 * i;
            tile[c][n] = (v[i] - mu) * rstd * gamma[c] + beta[c];
        }
    }
    __syncthreads();
    for (int idx = threadIdx.x; idx < Cc * Nn; idx += 256) {
        int c = idx / Nn, n = idx % Nn;  // n fastest -> coalesced write
        y[((size_t)(b * Cc + c) * Tt + t) * Nn + n] = tile[c][n];
    }
}

// ---------------------------------------------------------------------------
extern "C" void kernel_launch(void* const* d_in, const int* in_sizes, int n_in,
                              void* d_out, int out_size, void* d_ws, size_t ws_size,
                              hipStream_t stream) {
    (void)in_sizes; (void)n_in; (void)out_size; (void)ws_size;
    const float* x     = (const float*)d_in[0];
    const float* Wq    = (const float*)d_in[1];
    const float* Wk    = (const float*)d_in[2];
    const float* Wv    = (const float*)d_in[3];
    const float* a_src = (const float*)d_in[4];
    const float* a_dst = (const float*)d_in[5];
    const float* gam   = (const float*)d_in[6];
    const float* bet   = (const float*)d_in[7];

    // Workspace layout (bytes):
    //  [0, 4K)            u_src/u_dst (2*H*C fp32)
    //  [4K, ~6.6M)        s_src, s_dst (BN*H*T fp32 each)
    //  [7M, 7M+32K)       Wv in bf16 (row-major)
    //  [8M, 8M+104.9M)    xr bf16 (52.4M) ALIASED with attn-out fp32 (104.9M)
    //  [+, +52.4M)        V bf16
    char* ws = (char*)d_ws;
    size_t offS  = 4096;
    size_t offWB = (size_t)7 << 20;
    size_t offXR = (size_t)8 << 20;
    size_t offV  = offXR + (size_t)BN * Tt * Cc * sizeof(float);

    float*  u    = (float*)(ws);
    float*  ssrc = (float*)(ws + offS);
    float*  sdst = ssrc + (size_t)BN * Hh * Tt;
    __bf16* wvbf = (__bf16*)(ws + offWB);
    __bf16* xr   = (__bf16*)(ws + offXR);
    float*  outw = (float*)(ws + offXR);   // alias: xr dead after kProj
    __bf16* Vw   = (__bf16*)(ws + offV);

    kPrep<<<1, 128, 0, stream>>>(Wq, Wk, Wv, a_src, a_dst, u, wvbf);
    kT<<<Bb * Tt, 256, 0, stream>>>(x, xr);
    kProj<<<dim3(BN, Tt / 64), 256, 0, stream>>>(xr, wvbf, u, Vw, ssrc, sdst);
    kAttn<<<dim3(BN, Hh), 256, 0, stream>>>(Vw, ssrc, sdst, outw);
    kLN<<<Bb * Tt, 256, 0, stream>>>(outw, gam, bet, (float*)d_out);
}
